// RNN_833223655577
// MI455X (gfx1250) — compile-verified, matrix operand order checked by reading, not run on hardware
//
#include <hip/hip_runtime.h>

// ---------------- types ----------------
typedef __attribute__((ext_vector_type(16))) _Float16 v16h;
typedef __attribute__((ext_vector_type(8)))  float    v8f;

#define H90  90
#define GPAD 96          // padded gate-block width (z|r|h each padded 90->96)
#define NT_G 18          // total gate n-tiles (3 blocks * 6 tiles)

// ---------------- d_ws layout (bytes; total ~225.6 KB) ----------------
#define OFF_W1F   0u          // 4*32 tiles * 512 halves   = 131072 B  (BN scale pre-folded)
#define OFF_KF    131072u     // 18 tiles * 1KB            =  18432 B
#define OFF_RF    149504u     // 54 tiles * 1KB            =  55296 B
#define OFF_W2F   204800u     // 12 tiles                  =  12288 B
#define OFF_W3F   217088u     // 2 tiles                   =   2048 B
#define OFF_T     221184u     // 512 f32 BN shift (b1/mean/beta folded)
#define OFF_BZ    223232u     // 96 f32 (padded to 512B each)
#define OFF_BR    223744u
#define OFF_BXH   224256u
#define OFF_BIH   224768u
#define OFF_B2P   225280u     // 64 f32
#define OFF_B3P   225536u     // 16 f32

// Fragment element decode: B-matrix f16 layout per CDNA5 ISA 7.12.2:
// lane L -> column n = nt*16 + L%16 ; half h -> k = kt*32 + (h/8)*16 + 8*(L/16) + h%8
__device__ __forceinline__ void fragdec(int idx, int NT, int& kt, int& nt, int& n, int& k) {
  int tile = idx >> 9, r = idx & 511, lane = r >> 4, h = r & 15;
  kt = tile / NT; nt = tile - kt * NT;
  n  = nt * 16 + (lane & 15);
  k  = kt * 32 + ((h >> 3) << 4) + ((lane >> 4) << 3) + (h & 7);
}

// ---------------- prep kernel: weights -> f16 fragment layout in d_ws ----------------
__global__ __launch_bounds__(256) void prep_kernel(
    const float* __restrict__ w1, const float* __restrict__ b1,
    const float* __restrict__ gam, const float* __restrict__ bet,
    const float* __restrict__ mu,  const float* __restrict__ var,
    const float* __restrict__ gk,  const float* __restrict__ rk,
    const float* __restrict__ gb,  const float* __restrict__ w2,
    const float* __restrict__ b2,  const float* __restrict__ w3,
    const float* __restrict__ b3,  char* __restrict__ ws)
{
  int id = blockIdx.x * 256 + threadIdx.x;
  _Float16* w1f = (_Float16*)(ws + OFF_W1F);
  _Float16* kf  = (_Float16*)(ws + OFF_KF);
  _Float16* rf  = (_Float16*)(ws + OFF_RF);
  _Float16* w2f = (_Float16*)(ws + OFF_W2F);
  _Float16* w3f = (_Float16*)(ws + OFF_W3F);
  float* tc  = (float*)(ws + OFF_T);
  float* BZ  = (float*)(ws + OFF_BZ);
  float* BR  = (float*)(ws + OFF_BR);
  float* BXH = (float*)(ws + OFF_BXH);
  float* BIH = (float*)(ws + OFF_BIH);
  float* b2p = (float*)(ws + OFF_B2P);
  float* b3p = (float*)(ws + OFF_B3P);

  if (id < 4 * 32 * 512) {                       // w1 [128,512] with BN scale folded in
    int kt, nt, n, k; fragdec(id, 32, kt, nt, n, k);
    float s = gam[n] * rsqrtf(var[n] + 1e-3f);
    w1f[id] = (_Float16)(w1[k * 512 + n] * s);
  }
  if (id < NT_G * 512) {                         // gru_kernel [32,270] -> cols z|r|h padded to 96
    int kt, nt, n, k; fragdec(id, NT_G, kt, nt, n, k);
    int blk = n / GPAD, g = n % GPAD;
    kf[id] = (_Float16)((g < H90) ? gk[k * 270 + blk * H90 + g] : 0.f);
  }
  if (id < 3 * NT_G * 512) {                     // rec_kernel [90,270] -> K padded 96, cols padded
    int kt, nt, n, k; fragdec(id, NT_G, kt, nt, n, k);
    int blk = n / GPAD, g = n % GPAD;
    rf[id] = (_Float16)((k < H90 && g < H90) ? rk[k * 270 + blk * H90 + g] : 0.f);
  }
  if (id < 12 * 512) {                           // w2 [90,60] -> 96x64
    int kt, nt, n, k; fragdec(id, 4, kt, nt, n, k);
    w2f[id] = (_Float16)((k < H90 && n < 60) ? w2[k * 60 + n] : 0.f);
  }
  if (id < 2 * 512) {                            // w3 [60,10] -> 64x16
    int kt, nt, n, k; fragdec(id, 1, kt, nt, n, k);
    w3f[id] = (_Float16)((k < 60 && n < 10) ? w3[k * 10 + n] : 0.f);
  }
  if (id < 512) {                                // BN shift (scale stays folded in w1f)
    float s = gam[id] * rsqrtf(var[id] + 1e-3f);
    tc[id] = (b1[id] - mu[id]) * s + bet[id];
  }
  if (id < GPAD) {                               // merged gate biases, padded layout
    float bz = 0.f, br = 0.f, bx = 0.f, bi = 0.f;
    if (id < H90) {
      bz = gb[id]        + gb[270 + id];
      br = gb[90 + id]   + gb[270 + 90 + id];
      bx = gb[180 + id];
      bi = gb[270 + 180 + id];
    }
    BZ[id] = bz; BR[id] = br; BXH[id] = bx; BIH[id] = bi;
  }
  if (id < 64) b2p[id] = (id < 60) ? b2[id] : 0.f;
  if (id < 16) b3p[id] = (id < 10) ? b3[id] : 0.f;
}

// ---------------- main fused kernel ----------------
#define ZV8 v8f{0.f,0.f,0.f,0.f,0.f,0.f,0.f,0.f}

__device__ __forceinline__ v8f wmma_f16(v16h a, v16h b, v8f c) {
  return __builtin_amdgcn_wmma_f32_16x16x32_f16(false, a, false, b, (short)0, c, false, false);
}
// A-matrix f16 fragment: lane holds row m=L%16; halves 0-7: k=ko..ko+7, 8-15: k=ko+16..+23
__device__ __forceinline__ v16h ldA(const _Float16* row, int ko) {
  v16h a;
#pragma unroll
  for (int i = 0; i < 8; i++) { a[i] = row[ko + i]; a[i + 8] = row[ko + 16 + i]; }
  return a;
}
__device__ __forceinline__ v16h ldB(const _Float16* base, int tile, int lane) {
  return *(const v16h*)(base + (tile << 9) + (lane << 4));
}
__device__ __forceinline__ float fsigmoid(float x) {
  return __builtin_amdgcn_rcpf(1.f + __expf(-x));
}
__device__ __forceinline__ float ftanh(float x) {
  float e2 = __expf(2.f * x);
  return (e2 - 1.f) * __builtin_amdgcn_rcpf(e2 + 1.f);
}

__global__ __launch_bounds__(256) void fused_kernel(
    const float* __restrict__ x,
    const _Float16* __restrict__ w1f, const _Float16* __restrict__ kfG,
    const _Float16* __restrict__ rfG, const _Float16* __restrict__ w2f,
    const _Float16* __restrict__ w3f,
    const float* __restrict__ tc,
    const float* __restrict__ BZ,  const float* __restrict__ BR,
    const float* __restrict__ BXH, const float* __restrict__ BIH,
    const float* __restrict__ b2p, const float* __restrict__ b3p,
    float* __restrict__ out)
{
  // 130 KB static LDS total (gfx1250: up to 320 KB per workgroup)
  __shared__ __align__(16) _Float16 wK[NT_G * 512];        // 18 KB gru_kernel frags
  __shared__ __align__(16) _Float16 wR[3 * NT_G * 512];    // 54 KB rec_kernel frags
  __shared__ __align__(16) float    tct[512];              //  2 KB BN shift
  __shared__ __align__(16) _Float16 act[8][16][32];        //  8 KB per-step activation slab
  __shared__ __align__(16) _Float16 hsl[8][16][GPAD];      // 24 KB h state slab
  __shared__ __align__(16) _Float16 o2s[8][16][64];        // 16 KB layer-2 slab
  __shared__ __align__(16) float    lgt[8][16][16];        //  8 KB logits

  const int tid = threadIdx.x;
  // cooperative copy of recurrent weights + BN shift into LDS (uint4 = 8 halves)
  {
    const uint4* s0 = (const uint4*)kfG; uint4* d0 = (uint4*)wK;
    for (int i = tid; i < NT_G * 512 / 8; i += 256) d0[i] = s0[i];
    const uint4* s1 = (const uint4*)rfG; uint4* d1 = (uint4*)wR;
    for (int i = tid; i < 3 * NT_G * 512 / 8; i += 256) d1[i] = s1[i];
    for (int i = tid; i < 512; i += 256) tct[i] = tc[i];
  }
  __syncthreads();

  const int lane = tid & 31, wv = tid >> 5;
  const int l16 = lane & 15, lhi = lane >> 4;
  const int rowbase = (blockIdx.x * 8 + wv) * 16;

  // zero this wave's h slab
  {
    uint4* p = (uint4*)&hsl[wv][0][0];
    uint4 z(0u, 0u, 0u, 0u);
    for (int i = lane; i < 16 * GPAD / 8; i += 32) p[i] = z;
  }

  // x A-fragments (K=128, f32 -> f16), kept in VGPRs for whole kernel
  const float* xrow = x + (size_t)(rowbase + l16) * 128;
  v16h ax[4];
#pragma unroll
  for (int kt = 0; kt < 4; kt++) {
    int ko = kt * 32 + lhi * 8;
    v16h a;
#pragma unroll
    for (int i = 0; i < 8; i++) {
      a[i]     = (_Float16)xrow[ko + i];
      a[i + 8] = (_Float16)xrow[ko + 16 + i];
    }
    ax[kt] = a;
  }

  // hoist gate biases (per-lane column constants)
  float gbz[6], gbr[6], gbx[6], gbi[6];
#pragma unroll
  for (int j = 0; j < 6; j++) {
    int c = j * 16 + l16;
    gbz[j] = BZ[c]; gbr[j] = BR[c]; gbx[j] = BXH[c]; gbi[j] = BIH[c];
  }

  v8f hf[6];                                    // h state as D-fragments
#pragma unroll
  for (int j = 0; j < 6; j++) hf[j] = ZV8;

#pragma unroll 1
  for (int t = 0; t < 16; t++) {
    // ---- stage A lazily: the 32 activation columns step t consumes (2 tiles) ----
    // issue all 8 B-fragment loads first so WMMAs overlap the global loads
    v16h bw[8];
#pragma unroll
    for (int u = 0; u < 2; u++)
#pragma unroll
      for (int kt = 0; kt < 4; kt++)
        bw[u * 4 + kt] = ldB(w1f, kt * 32 + (t * 2 + u), lane);
#pragma unroll
    for (int u = 0; u < 2; u++) {
      v8f acc = ZV8;
#pragma unroll
      for (int kt = 0; kt < 4; kt++)
        acc = wmma_f16(ax[kt], bw[u * 4 + kt], acc);
      float sh = tct[(t * 2 + u) * 16 + l16];    // BN shift from LDS (scale folded in w1f)
#pragma unroll
      for (int v = 0; v < 8; v++) {
        float hv = acc[v] + sh;
        hv = hv >= 0.f ? hv : 0.2f * hv;                   // leaky 0.2
        act[wv][v + 8 * lhi][u * 16 + l16] = (_Float16)hv; // D-frag -> slab
      }
    }
    __builtin_amdgcn_wave_barrier();
    // reload as A-fragments
    v16h axs = ldA(&act[wv][l16][0], lhi * 8);
    v16h ah[3];
#pragma unroll
    for (int kt = 0; kt < 3; kt++) ah[kt] = ldA(&hsl[wv][l16][0], kt * 32 + lhi * 8);
    __builtin_amdgcn_wave_barrier();

    // ---- gates, tile column j (z/r/h blocks are 6 tiles apart -> fragments align) ----
#pragma unroll
    for (int j = 0; j < 6; j++) {
      v8f miz = ZV8, mir = ZV8, mih = ZV8;
#pragma unroll
      for (int kt = 0; kt < 3; kt++) {
        miz = wmma_f16(ah[kt], ldB(wR, kt * NT_G +      j, lane), miz);
        mir = wmma_f16(ah[kt], ldB(wR, kt * NT_G +  6 + j, lane), mir);
        mih = wmma_f16(ah[kt], ldB(wR, kt * NT_G + 12 + j, lane), mih);
      }
      v8f mxz = wmma_f16(axs, ldB(wK,      j, lane), ZV8);
      v8f mxr = wmma_f16(axs, ldB(wK,  6 + j, lane), ZV8);
      v8f mxh = wmma_f16(axs, ldB(wK, 12 + j, lane), ZV8);
#pragma unroll
      for (int v = 0; v < 8; v++) {
        float z  = fsigmoid(mxz[v] + miz[v] + gbz[j]);
        float r  = fsigmoid(mxr[v] + mir[v] + gbr[j]);
        float hh = ftanh(mxh[v] + gbx[j] + r * (mih[v] + gbi[j]));
        float hn = z * hf[j][v] + (1.f - z) * hh;
        hf[j][v] = hn;
        hsl[wv][v + 8 * lhi][j * 16 + l16] = (_Float16)hn;
      }
    }
    __builtin_amdgcn_wave_barrier();
  }

  // ---- stage C: h @ w2 (leaky 0.5), @ w3, softmax ----
  v16h ahf[3];
#pragma unroll
  for (int kt = 0; kt < 3; kt++) ahf[kt] = ldA(&hsl[wv][l16][0], kt * 32 + lhi * 8);
#pragma unroll
  for (int nt = 0; nt < 4; nt++) {
    v8f acc = ZV8;
#pragma unroll
    for (int kt = 0; kt < 3; kt++)
      acc = wmma_f16(ahf[kt], ldB(w2f, kt * 4 + nt, lane), acc);
    float b = b2p[nt * 16 + l16];
#pragma unroll
    for (int v = 0; v < 8; v++) {
      float o = acc[v] + b;
      o = o >= 0.f ? o : 0.5f * o;
      o2s[wv][v + 8 * lhi][nt * 16 + l16] = (_Float16)o;
    }
  }
  __builtin_amdgcn_wave_barrier();
  v16h ao0 = ldA(&o2s[wv][l16][0], lhi * 8);
  v16h ao1 = ldA(&o2s[wv][l16][0], 32 + lhi * 8);
  v8f lg = wmma_f16(ao0, ldB(w3f, 0, lane), ZV8);
  lg     = wmma_f16(ao1, ldB(w3f, 1, lane), lg);
  float b3v = b3p[l16];
#pragma unroll
  for (int v = 0; v < 8; v++) lgt[wv][v + 8 * lhi][l16] = lg[v] + b3v;
  __builtin_amdgcn_wave_barrier();

  if (lane < 16) {                               // one row per lane
    float e[10], mx = -1e30f;
#pragma unroll
    for (int c = 0; c < 10; c++) { e[c] = lgt[wv][lane][c]; mx = e[c] > mx ? e[c] : mx; }
    float s = 0.f;
#pragma unroll
    for (int c = 0; c < 10; c++) { e[c] = __expf(e[c] - mx); s += e[c]; }
    float inv = __builtin_amdgcn_rcpf(s);
    float* op = out + (size_t)(rowbase + lane) * 10;
#pragma unroll
    for (int c = 0; c < 10; c++) op[c] = e[c] * inv;
  }
}

// ---------------- launch ----------------
extern "C" void kernel_launch(void* const* d_in, const int* in_sizes, int n_in,
                              void* d_out, int out_size, void* d_ws, size_t ws_size,
                              hipStream_t stream) {
  const float* x   = (const float*)d_in[0];
  const float* w1  = (const float*)d_in[1];
  const float* b1  = (const float*)d_in[2];
  const float* gam = (const float*)d_in[3];
  const float* bet = (const float*)d_in[4];
  const float* mu  = (const float*)d_in[5];
  const float* var = (const float*)d_in[6];
  const float* gk  = (const float*)d_in[7];
  const float* rk  = (const float*)d_in[8];
  const float* gb  = (const float*)d_in[9];
  const float* w2  = (const float*)d_in[10];
  const float* b2  = (const float*)d_in[11];
  const float* w3  = (const float*)d_in[12];
  const float* b3  = (const float*)d_in[13];
  char* ws = (char*)d_ws;   // needs ~226 KB

  prep_kernel<<<256, 256, 0, stream>>>(w1, b1, gam, bet, mu, var, gk, rk, gb,
                                       w2, b2, w3, b3, ws);

  const int rows   = in_sizes[0] / 128;   // 131072
  const int blocks = rows / 128;          // 8 waves * 16 rows per block
  fused_kernel<<<blocks, 256, 0, stream>>>(
      x,
      (const _Float16*)(ws + OFF_W1F), (const _Float16*)(ws + OFF_KF),
      (const _Float16*)(ws + OFF_RF),  (const _Float16*)(ws + OFF_W2F),
      (const _Float16*)(ws + OFF_W3F),
      (const float*)(ws + OFF_T),
      (const float*)(ws + OFF_BZ),  (const float*)(ws + OFF_BR),
      (const float*)(ws + OFF_BXH), (const float*)(ws + OFF_BIH),
      (const float*)(ws + OFF_B2P), (const float*)(ws + OFF_B3P),
      (float*)d_out);
}
